// BiLSTMCrf_81346680586985
// MI455X (gfx1250) — matrix operand order repaired and checked by reading, not compiled, not run on hardware
//
#include <hip/hip_runtime.h>
#include <hip/hip_bf16.h>
#include <math.h>

// ---------------------------------------------------------------------------
// BiLSTM-CRF forward for MI455X (gfx1250, wave32, WMMA, TDM)
//   sent[8192] -> emb[100000,300] gather -> bidirectional LSTM(H=512)
//   -> proj to 12 tags + log_softmax -> Viterbi decode
// Strategy:
//   * input projections (x @ Wih.T + b) hoisted out of the recurrence as one
//     big bf16 WMMA GEMM per direction (pre-packed A/B fragments)
//   * recurrence: 1 workgroup (32 waves) per direction; h@Whh.T done with
//     v_wmma_f32_16x16x32_bf16, h replicated across the 16 A rows (row 0 of D
//     is the GEMV result); Whh (2MB bf16/dir) stays resident in the 192MB L2
//   * per-step xproj row (8KB) staged into an LDS double buffer by the
//     Tensor Data Mover, overlapped with the WMMA GEMV (s_wait_tensorcnt)
//   * viterbi: single wave, backpointers held in LDS (96KB < 320KB)
// ---------------------------------------------------------------------------

typedef __bf16 bf16_t;
typedef __bf16 v16bf __attribute__((ext_vector_type(16)));
typedef __bf16 v8bf  __attribute__((ext_vector_type(8)));
typedef float  v8f   __attribute__((ext_vector_type(8)));
typedef unsigned int v4u __attribute__((ext_vector_type(4)));
typedef int    v4i   __attribute__((ext_vector_type(4)));
typedef int    v8i   __attribute__((ext_vector_type(8)));

#define T_LEN   8192
#define D_IN    300
#define KB_IH   10          // ceil(300/32), K padded to 320
#define H_DIM   512
#define KB_HH   16          // 512/32
#define G4H     2048        // 4*H
#define NTILES  128         // 2048/16
#define MTILES  512         // 8192/16
#define NTAG    12
#define START_TAG 10
#define STOP_TAG  11
#define NEG_VAL  (-10000.0f)
#define VCH      64         // viterbi feats staging chunk (steps)

// ---------------------------------------------------------------------------
// TDM: async-load one contiguous row of `nfloats` f32 from global into LDS.
// D# built per CDNA5 ISA (2D degenerate: 1 x nfloats,
// data_size=4B, no pad/iterate/multicast).  Tracked by TENSORcnt.
// ---------------------------------------------------------------------------
__device__ __forceinline__ void tdm_load_row(const float* gsrc, void* lds_dst,
                                             int nfloats) {
    unsigned long long ga = (unsigned long long)(uintptr_t)gsrc;
    unsigned int la = (unsigned int)(uintptr_t)lds_dst;   // LDS addr = addr[31:0]
    v4u g0;
    g0[0] = 1u;                                             // count=1 (valid D#)
    g0[1] = la;                                             // lds_addr
    g0[2] = (unsigned int)(ga & 0xffffffffu);               // global_addr[31:0]
    g0[3] = (unsigned int)((ga >> 32) & 0x01ffffffu)        // global_addr[56:32]
          | (2u << 30);                                     // type=2 ("image")
    v8i g1;
    g1[0] = (int)(2u << 16);                  // workgroup_mask=0, data_size=4B
    g1[1] = (int)((unsigned)nfloats << 16);   // tensor_dim0[15:0] @ bit48
    g1[2] = (int)(1u << 16);                  // tensor_dim0 hi=0 | tensor_dim1=1
    g1[3] = (int)((unsigned)nfloats << 16);   // tensor_dim1 hi=0 | tile_dim0
    g1[4] = 1;                                // tile_dim1=1, tile_dim2=0
    g1[5] = nfloats;                          // tensor_dim0_stride[31:0]
    g1[6] = 0;                                // stride hi, dim1_stride lo
    g1[7] = 0;
    v4i z4 = {0, 0, 0, 0};
#if defined(__clang_major__) && (__clang_major__ >= 23)
    v8i z8 = {0, 0, 0, 0, 0, 0, 0, 0};
    __builtin_amdgcn_tensor_load_to_lds(g0, g1, z4, z4, z8, 0);
#else
    __builtin_amdgcn_tensor_load_to_lds(g0, g1, z4, z4, 0);
#endif
}

// ---------------------------------------------------------------------------
// Pack gathered embeddings into WMMA A-fragment layout (bf16, K padded to 320)
// A 16-bit 16x32 layout: lanes 0-15: K {0..7,16..23}; lanes 16-31: K {8..15,24..31}
// ---------------------------------------------------------------------------
__global__ void pack_x_kernel(const int* __restrict__ sent,
                              const float* __restrict__ emb,
                              bf16_t* __restrict__ Ap) {
    int tid = blockIdx.x * blockDim.x + threadIdx.x;
    const int total = MTILES * KB_IH * 32;
    if (tid >= total) return;
    int lane  = tid & 31;
    int kb    = (tid >> 5) % KB_IH;
    int mtile = tid / (32 * KB_IH);
    int m     = mtile * 16 + (lane & 15);
    const float* e = emb + (long)sent[m] * D_IN;
    int ks  = kb * 32;
    int ofs = (lane < 16) ? 0 : 8;
    bf16_t* dst = Ap + ((size_t)(mtile * KB_IH + kb) * 32 + lane) * 16;
#pragma unroll
    for (int j = 0; j < 8; ++j) {
        int k0 = ks + ofs + j;          // first 8 halfs
        int k1 = ks + 16 + ofs + j;     // second 8 halfs
        dst[j]     = (bf16_t)((k0 < D_IN) ? e[k0] : 0.0f);
        dst[j + 8] = (bf16_t)((k1 < D_IN) ? e[k1] : 0.0f);
    }
}

// ---------------------------------------------------------------------------
// Pack a weight matrix W [2048 rows x K cols] (we need W.T as K x N B-matrix)
// into WMMA B-fragment layout (bf16). B 16-bit 32x16 layout:
// lane n<16 holds K {0..7,16..23} of column n; lane n+16 holds K {8..15,24..31}.
// ---------------------------------------------------------------------------
__global__ void pack_w_kernel(const float* __restrict__ W, int K, int KB,
                              bf16_t* __restrict__ Bp) {
    int tid = blockIdx.x * blockDim.x + threadIdx.x;
    int total = NTILES * KB * 32;
    if (tid >= total) return;
    int lane  = tid & 31;
    int kb    = (tid >> 5) % KB;
    int ntile = tid / (32 * KB);
    int n     = ntile * 16 + (lane & 15);
    int kbase = kb * 32 + ((lane < 16) ? 0 : 8);
    bf16_t* dst = Bp + ((size_t)(ntile * KB + kb) * 32 + lane) * 16;
#pragma unroll
    for (int j = 0; j < 16; ++j) {
        int k = kbase + ((j < 8) ? j : (8 + j));
        dst[j] = (bf16_t)((k < K) ? W[(size_t)n * K + k] : 0.0f);
    }
}

// ---------------------------------------------------------------------------
// Input projection GEMM: xp[T,2048] = X[T,320] @ Wih.T (packed) + bias
// One wave per 16x16 output tile.  Pure b128 loads + v_wmma.
// ---------------------------------------------------------------------------
__global__ void input_proj_gemm(const bf16_t* __restrict__ Ap,
                                const bf16_t* __restrict__ Bp,
                                const float* __restrict__ bias,
                                float* __restrict__ xp) {
    int lane  = threadIdx.x & 31;
    int gw    = blockIdx.x * (blockDim.x >> 5) + (threadIdx.x >> 5);
    int mtile = gw >> 7;          // / 128
    int ntile = gw & 127;
    v8f acc = {0.f, 0.f, 0.f, 0.f, 0.f, 0.f, 0.f, 0.f};
#pragma unroll
    for (int kb = 0; kb < KB_IH; ++kb) {
        v16bf a = *(const v16bf*)(Ap + ((size_t)(mtile * KB_IH + kb) * 32 + lane) * 16);
        v16bf b = *(const v16bf*)(Bp + ((size_t)(ntile * KB_IH + kb) * 32 + lane) * 16);
        acc = __builtin_amdgcn_wmma_f32_16x16x32_bf16(false, a, false, b,
                                                      (short)0, acc, false, false);
    }
    // D layout: VGPR r -> (M=r, N=lane) lanes 0-15 ; (M=8+r, N=lane-16) lanes 16-31
    int col   = ntile * 16 + (lane & 15);
    int rbase = mtile * 16 + ((lane < 16) ? 0 : 8);
    float bcol = bias[col];
#pragma unroll
    for (int r = 0; r < 8; ++r)
        xp[(size_t)(rbase + r) * G4H + col] = acc[r] + bcol;
}

__device__ __forceinline__ float sigm(float x) {
    return 1.0f / (1.0f + __expf(-x));
}

// ---------------------------------------------------------------------------
// Sequential LSTM recurrence, one block (32 waves) per direction (grid = 2).
// Per step: g[2048] = xstage[t] + h @ Whh.T  via WMMA with h replicated into
// all 16 A-rows; then gate math on 512 threads.  The xproj row for step s+1
// is DMA'd into LDS by the TDM while step s+1's GEMV runs.
// ---------------------------------------------------------------------------
__global__ void lstm_recurrent_kernel(const bf16_t* __restrict__ BhhF,
                                      const bf16_t* __restrict__ BhhB,
                                      const float*  __restrict__ xpF,
                                      const float*  __restrict__ xpB,
                                      bf16_t* __restrict__ hF,
                                      bf16_t* __restrict__ hB) {
    const int dir = blockIdx.x;
    const bf16_t* Bp   = dir ? BhhB : BhhF;
    const float*  xp   = dir ? xpB  : xpF;
    bf16_t*       hout = dir ? hB   : hF;

    __shared__ __align__(16) float  xstage[2][G4H];   // TDM double buffer (16KB)
    __shared__ __align__(16) float  gacc[G4H];
    __shared__ __align__(16) float  cstate[H_DIM];
    __shared__ __align__(16) bf16_t hbf[H_DIM];

    const int tid  = threadIdx.x;
    const int lane = tid & 31;
    const int w    = tid >> 5;
    const int ofs  = (lane < 16) ? 0 : 8;

    if (tid < H_DIM) { cstate[tid] = 0.0f; hbf[tid] = (bf16_t)0.0f; }

    // Prologue: stage xproj row for step 0 (wave 31 issues the DMA).
    if (w == 31) {
        int trow0 = dir ? (T_LEN - 1) : 0;
        tdm_load_row(xp + (size_t)trow0 * G4H, &xstage[0][0], G4H);
    }
    __syncthreads();

    const size_t ntStride = (size_t)KB_HH * 32 * 16;   // halfs between n-tiles

    for (int s = 0; s < T_LEN; ++s) {
        v8f a0 = {0.f,0.f,0.f,0.f,0.f,0.f,0.f,0.f};
        v8f a1 = a0, a2 = a0, a3 = a0;
#pragma unroll 4
        for (int kb = 0; kb < KB_HH; ++kb) {
            // Build replicated-h A fragment from LDS (2x ds_load_b128)
            int base = kb * 32;
            v8bf lo = *(const v8bf*)&hbf[base + ofs];
            v8bf hi = *(const v8bf*)&hbf[base + 16 + ofs];
            v16bf a;
#pragma unroll
            for (int j = 0; j < 8; ++j) { a[j] = lo[j]; a[j + 8] = hi[j]; }

            const bf16_t* bptr = Bp + ((size_t)((w * 4) * KB_HH + kb) * 32 + lane) * 16;
            __builtin_prefetch(bptr + 512, 0, 1);   // next kb block (global_prefetch)
            v16bf b0 = *(const v16bf*)(bptr);
            v16bf b1 = *(const v16bf*)(bptr + ntStride);
            v16bf b2 = *(const v16bf*)(bptr + 2 * ntStride);
            v16bf b3 = *(const v16bf*)(bptr + 3 * ntStride);
            a0 = __builtin_amdgcn_wmma_f32_16x16x32_bf16(false, a, false, b0, (short)0, a0, false, false);
            a1 = __builtin_amdgcn_wmma_f32_16x16x32_bf16(false, a, false, b1, (short)0, a1, false, false);
            a2 = __builtin_amdgcn_wmma_f32_16x16x32_bf16(false, a, false, b2, (short)0, a2, false, false);
            a3 = __builtin_amdgcn_wmma_f32_16x16x32_bf16(false, a, false, b3, (short)0, a3, false, false);
        }
        // All 16 D rows identical (A rows replicated); row 0 lives in
        // acc element 0, lanes 0-15 (N = lane).
        if (lane < 16) {
            gacc[(w * 4 + 0) * 16 + lane] = a0[0];
            gacc[(w * 4 + 1) * 16 + lane] = a1[0];
            gacc[(w * 4 + 2) * 16 + lane] = a2[0];
            gacc[(w * 4 + 3) * 16 + lane] = a3[0];
        }
        // Ensure this step's staged xproj row has landed in LDS.
        if (w == 31) __builtin_amdgcn_s_wait_tensorcnt(0);
        __syncthreads();

        if (tid < H_DIM) {
            const float* xr = xstage[s & 1];
            float gi = xr[tid]             + gacc[tid];
            float gf = xr[H_DIM + tid]     + gacc[H_DIM + tid];
            float gg = xr[2 * H_DIM + tid] + gacc[2 * H_DIM + tid];
            float go = xr[3 * H_DIM + tid] + gacc[3 * H_DIM + tid];
            float c  = sigm(gf) * cstate[tid] + sigm(gi) * tanhf(gg);
            float h  = sigm(go) * tanhf(c);
            int trow = dir ? (T_LEN - 1 - s) : s;
            cstate[tid] = c;
            hbf[tid]    = (bf16_t)h;
            hout[(size_t)trow * H_DIM + tid] = (bf16_t)h;
        }
        __syncthreads();

        // Kick off the DMA for the next step's xproj row; it overlaps the
        // whole next GEMV.  Buffer (s+1)&1 was last read two barriers ago.
        if (w == 31 && (s + 1) < T_LEN) {
            int trow1 = dir ? (T_LEN - 2 - s) : (s + 1);
            tdm_load_row(xp + (size_t)trow1 * G4H, &xstage[(s + 1) & 1][0], G4H);
        }
    }
}

// ---------------------------------------------------------------------------
// feats[t, tag] = log_softmax( [hf|hb] @ Wproj.T + bproj )
// 64 threads / block, 4 timesteps / block, 12 active lanes per timestep.
// ---------------------------------------------------------------------------
__global__ void proj_logsoftmax_kernel(const bf16_t* __restrict__ hF,
                                       const bf16_t* __restrict__ hB,
                                       const float* __restrict__ Wp,
                                       const float* __restrict__ bp,
                                       float* __restrict__ feats) {
    __shared__ float logits[4][16];
    int tid = threadIdx.x;
    int tl  = tid >> 4;
    int tg  = tid & 15;
    int t   = blockIdx.x * 4 + tl;
    float l = 0.0f;
    if (tg < NTAG) {
        const float* wrow = Wp + (size_t)tg * (2 * H_DIM);
        const bf16_t* hf = hF + (size_t)t * H_DIM;
        const bf16_t* hb = hB + (size_t)t * H_DIM;
        float acc = bp[tg];
        for (int k = 0; k < H_DIM; ++k) acc += wrow[k] * (float)hf[k];
        for (int k = 0; k < H_DIM; ++k) acc += wrow[H_DIM + k] * (float)hb[k];
        l = acc;
    }
    logits[tl][tg] = l;
    __syncthreads();
    if (tg < NTAG) {
        float m = -3.4e38f;
        for (int i = 0; i < NTAG; ++i) m = fmaxf(m, logits[tl][i]);
        float ssum = 0.0f;
        for (int i = 0; i < NTAG; ++i) ssum += __expf(logits[tl][i] - m);
        feats[(size_t)t * NTAG + tg] = l - m - __logf(ssum);
    }
}

// ---------------------------------------------------------------------------
// Viterbi: single wave32. Backpointers (96KB) + feats chunks in LDS.
// out[0] = score, out[1..T] = tags (as float).
// ---------------------------------------------------------------------------
__global__ void viterbi_kernel(const float* __restrict__ feats,
                               const float* __restrict__ trans,
                               float* __restrict__ out) {
    extern __shared__ char smem[];
    float* trans_s = (float*)smem;                       // 144 f32
    float* v_s     = trans_s + 144;                      // 16 f32
    float* fin     = v_s + 16;                           // 16 f32
    float* fch     = fin + 16;                           // VCH*12 f32
    unsigned char* ptrs = (unsigned char*)(fch + VCH * NTAG);   // T*12 u8

    int lane = threadIdx.x;
    for (int i = lane; i < NTAG * NTAG; i += 32) trans_s[i] = trans[i];
    if (lane < 16) v_s[lane] = (lane == START_TAG) ? 0.0f : NEG_VAL;
    __syncthreads();

    for (int t = 0; t < T_LEN; ++t) {
        if ((t & (VCH - 1)) == 0) {
            __syncthreads();
            for (int i = lane; i < VCH * NTAG; i += 32)
                fch[i] = feats[(size_t)t * NTAG + i];
            __syncthreads();
        }
        if (lane < NTAG) {
            float best = -3.4e38f; int bi = 0;
#pragma unroll
            for (int i = 0; i < NTAG; ++i) {
                float vv = v_s[i] + trans_s[i * NTAG + lane];
                if (vv > best) { best = vv; bi = i; }
            }
            ptrs[(size_t)t * NTAG + lane] = (unsigned char)bi;
            fin[lane] = best + fch[(t & (VCH - 1)) * NTAG + lane];
        }
        __syncthreads();
        if (lane < NTAG) v_s[lane] = fin[lane];
        __syncthreads();
    }

    if (lane < NTAG) fin[lane] = v_s[lane] + trans_s[lane * NTAG + STOP_TAG];
    __syncthreads();
    if (lane == 0) {
        float best = -3.4e38f; int bj = 0;
        for (int j = 0; j < NTAG; ++j)
            if (fin[j] > best) { best = fin[j]; bj = j; }
        out[0] = best;
        int y = bj;
        out[1 + (T_LEN - 1)] = (float)y;        // tags[T-1] = best
        for (int t = T_LEN - 1; t >= 1; --t) {
            y = ptrs[(size_t)t * NTAG + y];
            out[t] = (float)y;                  // tags[t-1]
        }
    }
}

// ---------------------------------------------------------------------------
extern "C" void kernel_launch(void* const* d_in, const int* in_sizes, int n_in,
                              void* d_out, int out_size, void* d_ws, size_t ws_size,
                              hipStream_t stream) {
    const int*   sent  = (const int*)d_in[0];
    const float* emb   = (const float*)d_in[1];
    const float* WihF  = (const float*)d_in[2];
    const float* WhhF  = (const float*)d_in[3];
    const float* bF    = (const float*)d_in[4];
    const float* WihB  = (const float*)d_in[5];
    const float* WhhB  = (const float*)d_in[6];
    const float* bB    = (const float*)d_in[7];
    const float* Wp    = (const float*)d_in[8];
    const float* bp    = (const float*)d_in[9];
    const float* trans = (const float*)d_in[10];
    float* out = (float*)d_out;

    char* ws = (char*)d_ws;
    size_t off = 0;
    auto alloc = [&](size_t bytes) -> void* {
        void* p = ws + off;
        off = (off + bytes + 255) & ~(size_t)255;
        return p;
    };

    bf16_t* Ap    = (bf16_t*)alloc((size_t)MTILES * KB_IH * 32 * 16 * sizeof(bf16_t));
    bf16_t* BihFp = (bf16_t*)alloc((size_t)NTILES * KB_IH * 32 * 16 * sizeof(bf16_t));
    bf16_t* BihBp = (bf16_t*)alloc((size_t)NTILES * KB_IH * 32 * 16 * sizeof(bf16_t));
    bf16_t* BhhFp = (bf16_t*)alloc((size_t)NTILES * KB_HH * 32 * 16 * sizeof(bf16_t));
    bf16_t* BhhBp = (bf16_t*)alloc((size_t)NTILES * KB_HH * 32 * 16 * sizeof(bf16_t));
    float*  xpF   = (float*)alloc((size_t)T_LEN * G4H * sizeof(float));
    float*  xpB   = (float*)alloc((size_t)T_LEN * G4H * sizeof(float));
    bf16_t* hFp   = (bf16_t*)alloc((size_t)T_LEN * H_DIM * sizeof(bf16_t));
    bf16_t* hBp   = (bf16_t*)alloc((size_t)T_LEN * H_DIM * sizeof(bf16_t));
    float*  featp = (float*)alloc((size_t)T_LEN * NTAG * sizeof(float));

    const int thr = 256;
    // 1) gather + pack A fragments (bf16, K padded 300->320)
    pack_x_kernel<<<(MTILES * KB_IH * 32 + thr - 1) / thr, thr, 0, stream>>>(sent, emb, Ap);
    // 2) pack weight B fragments
    pack_w_kernel<<<(NTILES * KB_IH * 32 + thr - 1) / thr, thr, 0, stream>>>(WihF, D_IN, KB_IH, BihFp);
    pack_w_kernel<<<(NTILES * KB_IH * 32 + thr - 1) / thr, thr, 0, stream>>>(WihB, D_IN, KB_IH, BihBp);
    pack_w_kernel<<<(NTILES * KB_HH * 32 + thr - 1) / thr, thr, 0, stream>>>(WhhF, H_DIM, KB_HH, BhhFp);
    pack_w_kernel<<<(NTILES * KB_HH * 32 + thr - 1) / thr, thr, 0, stream>>>(WhhB, H_DIM, KB_HH, BhhBp);
    // 3) input projection GEMMs (one wave per 16x16 tile, 8 waves/block)
    input_proj_gemm<<<(MTILES * NTILES) / 8, 256, 0, stream>>>(Ap, BihFp, bF, xpF);
    input_proj_gemm<<<(MTILES * NTILES) / 8, 256, 0, stream>>>(Ap, BihBp, bB, xpB);
    // 4) sequential bidirectional recurrence (2 WGPs, one per direction)
    lstm_recurrent_kernel<<<2, 1024, 0, stream>>>(BhhFp, BhhBp, xpF, xpB, hFp, hBp);
    // 5) projection + log_softmax
    proj_logsoftmax_kernel<<<T_LEN / 4, 64, 0, stream>>>(hFp, hBp, Wp, bp, featp);
    // 6) viterbi decode (backpointers in LDS: 96KB + staging)
    size_t smem = (size_t)(144 + 16 + 16 + VCH * NTAG) * sizeof(float)
                + (size_t)T_LEN * NTAG;
    viterbi_kernel<<<1, 32, smem, stream>>>(featp, trans, out);
}
